// sLSTMLayer_31473520345368
// MI455X (gfx1250) — matrix operand order repaired and checked by reading, not compiled, not run on hardware
//
#include <hip/hip_runtime.h>
#include <hip/hip_bf16.h>
#include <stdint.h>

// ---------------------------------------------------------------------------
// sLSTM layer: B=4096, IN=1024, H=1024
//   gate_g = [x|h_prev] @ [Wg_x|Wg_h]^T + b_g      (K = 2048, bf16 WMMA, f32 acc)
//   then stabilized sLSTM pointwise update in place over d_out.
// Double-buffered LDS pipeline, 64x64 per-wave tile (16 WMMA per K-step).
// ---------------------------------------------------------------------------

#define B_DIM  4096
#define H_DIM  1024
#define IN_DIM 1024
#define K_DIM  2048   // IN + H fused

#define BM 256
#define BN 128
#define BK 32
#define NK (K_DIM / BK)
#define LDS_STRIDE 40   // halves per LDS row (32 data + 8 pad): 80B rows, 16B aligned, conflict-free

typedef __attribute__((ext_vector_type(16))) __bf16 v16bf;
typedef __attribute__((ext_vector_type(8)))  float  v8f;

union Frag { v16bf v; uint4 q[2]; };

__device__ __forceinline__ unsigned short f2bf(float f) {
  union { float f; uint32_t u; } v; v.f = f;
  uint32_t u = v.u;
  uint32_t r = u + 0x7FFFu + ((u >> 16) & 1u);   // round-to-nearest-even
  return (unsigned short)(r >> 16);
}

// ---- pack [x | h_prev] -> bf16 A (4096 x 2048), 4 elems / thread ----------
__global__ void pack_A(const float* __restrict__ x, const float* __restrict__ h,
                       unsigned short* __restrict__ A) {
  int e = (blockIdx.x * blockDim.x + threadIdx.x) * 4;
  int row = e >> 11;          // / 2048
  int col = e & 2047;
  const float* src = (col < IN_DIM) ? (x + row * IN_DIM + col)
                                    : (h + row * H_DIM + (col - IN_DIM));
  float4 f = *(const float4*)src;
  ushort4 o;
  o.x = f2bf(f.x); o.y = f2bf(f.y); o.z = f2bf(f.z); o.w = f2bf(f.w);
  *(ushort4*)(A + e) = o;
}

// ---- pack 8 weights -> 4 fused bf16 W_g (1024 x 2048 each) ----------------
__global__ void pack_W(const float* __restrict__ Wi_x, const float* __restrict__ Wi_h,
                       const float* __restrict__ Wf_x, const float* __restrict__ Wf_h,
                       const float* __restrict__ Wz_x, const float* __restrict__ Wz_h,
                       const float* __restrict__ Wo_x, const float* __restrict__ Wo_h,
                       unsigned short* __restrict__ W) {
  int e = (blockIdx.x * blockDim.x + threadIdx.x) * 4;
  int g = e >> 21;                       // / (1024*2048)
  int r = e & ((1 << 21) - 1);
  int n = r >> 11;
  int col = r & 2047;
  const float* wx; const float* wh;
  switch (g) {
    case 0: wx = Wi_x; wh = Wi_h; break;
    case 1: wx = Wf_x; wh = Wf_h; break;
    case 2: wx = Wz_x; wh = Wz_h; break;
    default: wx = Wo_x; wh = Wo_h; break;
  }
  const float* src = (col < IN_DIM) ? (wx + n * IN_DIM + col)
                                    : (wh + n * H_DIM + (col - IN_DIM));
  float4 f = *(const float4*)src;
  ushort4 o;
  o.x = f2bf(f.x); o.y = f2bf(f.y); o.z = f2bf(f.z); o.w = f2bf(f.w);
  *(ushort4*)(W + e) = o;
}

// ---- fused-K GEMM: out[g] = A @ W_g^T + b_g  (bf16 WMMA, f32 accum) -------
// Block tile 256x128 (8 waves, each 64x64 = 16 WMMA tiles). Double-buffered LDS.
__global__ __launch_bounds__(256)
void gemm_gates(const unsigned short* __restrict__ A,
                const unsigned short* __restrict__ W,
                const float* __restrict__ bi, const float* __restrict__ bfv,
                const float* __restrict__ bz, const float* __restrict__ bo,
                float* __restrict__ out) {
  __shared__ unsigned short As[2][BM * LDS_STRIDE];   // 2 x 20 KB
  __shared__ unsigned short Ws[2][BN * LDS_STRIDE];   // 2 x 10 KB

  const int gate = blockIdx.z;
  const int bm = blockIdx.y * BM;
  const int bn = blockIdx.x * BN;
  const unsigned short* __restrict__ Wg = W + (size_t)gate * H_DIM * K_DIM;
  const float* __restrict__ bias =
      (gate == 0) ? bi : (gate == 1) ? bfv : (gate == 2) ? bz : bo;
  float* __restrict__ outg = out + (size_t)gate * B_DIM * H_DIM;

  const int tid   = threadIdx.x;
  const int lane  = tid & 31;
  const int wave  = tid >> 5;
  const int waveM = wave >> 1;   // 4 waves along M: 64 rows each
  const int waveN = wave & 1;    // 2 waves along N: 64 cols each
  const int laneM = lane & 15;
  const int hiSel = lane >> 4;   // 0: lanes 0-15, 1: lanes 16-31

  const v8f vzero = {0.f, 0.f, 0.f, 0.f, 0.f, 0.f, 0.f, 0.f};
  v8f acc[4][4];
#pragma unroll
  for (int mt = 0; mt < 4; ++mt)
#pragma unroll
    for (int nt = 0; nt < 4; ++nt) acc[mt][nt] = vzero;

  // Global staging pointers: A -> one 32-half row per thread (64B);
  //                          W -> half-row (32B) per thread.
  const unsigned short* gA = A  + (size_t)(bm + tid) * K_DIM;
  const unsigned short* gW = Wg + (size_t)(bn + (tid >> 1)) * K_DIM + (tid & 1) * 16;

  uint4 aReg[4], wReg[2];
#define LOAD_GLOBAL(K0)                                                     \
  { const uint4* ga = (const uint4*)(gA + (K0));                            \
    aReg[0] = ga[0]; aReg[1] = ga[1]; aReg[2] = ga[2]; aReg[3] = ga[3];     \
    const uint4* gw = (const uint4*)(gW + (K0));                            \
    wReg[0] = gw[0]; wReg[1] = gw[1]; }

#define STORE_LDS(BUF)                                                      \
  { unsigned short* da = As[BUF] + tid * LDS_STRIDE;                        \
    ((uint4*)da)[0] = aReg[0]; ((uint4*)da)[1] = aReg[1];                   \
    ((uint4*)da)[2] = aReg[2]; ((uint4*)da)[3] = aReg[3];                   \
    unsigned short* dw = Ws[BUF] + (tid >> 1) * LDS_STRIDE + (tid & 1) * 16;\
    ((uint4*)dw)[0] = wReg[0]; ((uint4*)dw)[1] = wReg[1]; }

  // prologue: tile 0 into buffer 0
  LOAD_GLOBAL(0)
  STORE_LDS(0)
  __syncthreads();

  for (int kt = 0; kt < NK; ++kt) {
    const int cur = kt & 1;

    // issue next tile's global loads early: latency hides under WMMA below
    if (kt + 1 < NK) {
      LOAD_GLOBAL((kt + 1) * BK)
      if (kt + 2 < NK) {
        __builtin_prefetch((const void*)(gA + (kt + 2) * BK), 0, 3);
        __builtin_prefetch((const void*)(gW + (kt + 2) * BK), 0, 3);
      }
    }

    // ---- compute from buffer `cur` ----
    const unsigned short* aBase = As[cur];
    const unsigned short* wBase = Ws[cur];

    // B fragments: 32x16 tile; lane L -> col N=L%16, K chunk hiSel*16 (32B contiguous)
    Frag bfr[4];
#pragma unroll
    for (int nt = 0; nt < 4; ++nt) {
      const unsigned short* p =
          wBase + (waveN * 64 + nt * 16 + laneM) * LDS_STRIDE + hiSel * 16;
      bfr[nt].q[0] = *(const uint4*)p;
      bfr[nt].q[1] = *(const uint4*)(p + 8);
    }
    // A fragments: 16x32 tile; lane L -> row M=L%16, K chunk hiSel*16 (32B contiguous)
#pragma unroll
    for (int mt = 0; mt < 4; ++mt) {
      Frag af;
      const unsigned short* p =
          aBase + (waveM * 64 + mt * 16 + laneM) * LDS_STRIDE + hiSel * 16;
      af.q[0] = *(const uint4*)p;
      af.q[1] = *(const uint4*)(p + 8);
#pragma unroll
      for (int nt = 0; nt < 4; ++nt)
        acc[mt][nt] = __builtin_amdgcn_wmma_f32_16x16x32_bf16(
            false, af.v, false, bfr[nt].v, (short)0, acc[mt][nt],
            false, false);
    }

    // stage next tile into the other buffer (safe: last read of it was kt-1,
    // protected by the barrier at the end of kt-1)
    if (kt + 1 < NK) STORE_LDS(1 - cur)
    __syncthreads();
  }

  // epilogue: bias + store. C/D layout: lane L -> N=L%16; VGPR r -> M=r+hiSel*8
#pragma unroll
  for (int nt = 0; nt < 4; ++nt) {
    const int col = bn + waveN * 64 + nt * 16 + laneM;
    const float bv = bias[col];
#pragma unroll
    for (int mt = 0; mt < 4; ++mt) {
      const int rowBase = bm + waveM * 64 + mt * 16 + hiSel * 8;
#pragma unroll
      for (int r = 0; r < 8; ++r)
        outg[(size_t)(rowBase + r) * H_DIM + col] = acc[mt][nt][r] + bv;
    }
  }
#undef LOAD_GLOBAL
#undef STORE_LDS
}

// ---- pointwise sLSTM update, in place over d_out --------------------------
__global__ void slstm_pointwise(float* __restrict__ out,
                                const float* __restrict__ c_prev,
                                const float* __restrict__ n_prev,
                                const float* __restrict__ m_prev) {
  const int BH = B_DIM * H_DIM;
  int e = blockIdx.x * blockDim.x + threadIdx.x;
  float it = out[e];
  float fa = out[BH + e];
  float za = out[2 * BH + e];
  float oa = out[3 * BH + e];

  float sf    = 1.f / (1.f + __expf(-fa));
  float log_f = __logf(fmaxf(sf, 1e-8f));
  float mp    = m_prev[e];
  float m_t   = fmaxf(log_f + mp, it);
  float ip    = __expf(it - m_t);
  float fp    = __expf(log_f + mp - m_t);
  float z     = tanhf(za);
  float c_t   = fp * c_prev[e] + ip * z;
  float n_t   = fp * n_prev[e] + ip;
  float so    = 1.f / (1.f + __expf(-oa));
  float h_t   = so * (c_t / fmaxf(n_t, 1e-6f));

  out[e]          = h_t;
  out[BH + e]     = c_t;
  out[2 * BH + e] = n_t;
  out[3 * BH + e] = m_t;
}

// ---------------------------------------------------------------------------
extern "C" void kernel_launch(void* const* d_in, const int* in_sizes, int n_in,
                              void* d_out, int out_size, void* d_ws, size_t ws_size,
                              hipStream_t stream) {
  const float* x      = (const float*)d_in[0];
  const float* h_prev = (const float*)d_in[1];
  const float* c_prev = (const float*)d_in[2];
  const float* n_prev = (const float*)d_in[3];
  const float* m_prev = (const float*)d_in[4];
  const float* Wi_x = (const float*)d_in[5];
  const float* bi   = (const float*)d_in[6];
  const float* Wi_h = (const float*)d_in[7];
  const float* Wf_x = (const float*)d_in[8];
  const float* bf   = (const float*)d_in[9];
  const float* Wf_h = (const float*)d_in[10];
  const float* Wz_x = (const float*)d_in[11];
  const float* bz   = (const float*)d_in[12];
  const float* Wz_h = (const float*)d_in[13];
  const float* Wo_x = (const float*)d_in[14];
  const float* bo   = (const float*)d_in[15];
  const float* Wo_h = (const float*)d_in[16];

  unsigned short* Abf = (unsigned short*)d_ws;                 // 4096x2048 bf16 (16 MB)
  unsigned short* Wbf = Abf + (size_t)B_DIM * K_DIM;           // 4x1024x2048 bf16 (16 MB)

  pack_A<<<(B_DIM * K_DIM / 4) / 256, 256, 0, stream>>>(x, h_prev, Abf);
  pack_W<<<(4 * H_DIM * K_DIM / 4) / 256, 256, 0, stream>>>(
      Wi_x, Wi_h, Wf_x, Wf_h, Wz_x, Wz_h, Wo_x, Wo_h, Wbf);

  dim3 grid(H_DIM / BN, B_DIM / BM, 4);
  gemm_gates<<<grid, 256, 0, stream>>>(Abf, Wbf, bi, bf, bz, bo, (float*)d_out);

  slstm_pointwise<<<(B_DIM * H_DIM) / 256, 256, 0, stream>>>(
      (float*)d_out, c_prev, n_prev, m_prev);
}
